// CausalSelfAttention_35433480192102
// MI455X (gfx1250) — compile-verified, matrix operand order checked by reading, not run on hardware
//
#include <hip/hip_runtime.h>
#include <math.h>

// ---------------- Types ----------------
typedef __attribute__((ext_vector_type(16))) __bf16 v16bf;
typedef __attribute__((ext_vector_type(8)))  __bf16 v8bf;
typedef __attribute__((ext_vector_type(8)))  float  v8f;
typedef __attribute__((ext_vector_type(4)))  unsigned v4u;
typedef __attribute__((ext_vector_type(4)))  int    v4i;
typedef __attribute__((ext_vector_type(8)))  int    v8i;

#define N_B 4
#define N_T 2048
#define N_C 2048
#define N_H 16
#define N_HD 128
#define LDT 48           // GEMM LDS row stride (bf16): 96B = 64B row + 32B TDM pad
#define KST 136          // flash K-tile LDS row stride (bf16): 272B = 256B + 16B TDM pad
#define VST 72           // flash V^T LDS row stride (bf16)
#define GEMM_BUF 24576u  // bytes per double-buffer slot (A tile 12288 + B tile 12288)

#if defined(__has_builtin)
#if __has_builtin(__builtin_amdgcn_tensor_load_to_lds)
#define HAVE_TDM 1
#endif
#endif
#ifndef HAVE_TDM
#define HAVE_TDM 0
#endif

__device__ __forceinline__ v8f wmma_bf16(v16bf a, v16bf b, v8f c) {
  return __builtin_amdgcn_wmma_f32_16x16x32_bf16(false, a, false, b, (short)0, c, false, false);
}

// Per-lane fragment slice: 8 contiguous bf16 at p0, 8 at p1 (ISA 16-bit A/B layout).
__device__ __forceinline__ v16bf ldfrag(const __bf16* p0, const __bf16* p1) {
  union { v16bf f; v8bf h[2]; } u;
  u.h[0] = *(const v8bf*)p0;
  u.h[1] = *(const v8bf*)p1;
  return u.f;
}

#if HAVE_TDM
// Tensor DMA descriptor for a 2D tile of 2-byte elements (ISA 8.3/8.4).
// Built uniformly (all waves) so it stays scalar; issued by one wave.
struct TdmDesc { v4u g0; v8i g1; };

__device__ __forceinline__ TdmDesc tdm_make_2d(unsigned lds_off, const void* gaddr,
                                               unsigned tile_d0, unsigned tile_d1,
                                               unsigned tensor_d0, unsigned tensor_d1,
                                               unsigned stride_d0,
                                               unsigned pad_int_code, unsigned pad_amt_code,
                                               bool pad_en) {
  unsigned long long ga = (unsigned long long)gaddr;
  TdmDesc d;
  d.g0[0] = 1u;                                              // count=1, user-mode load
  d.g0[1] = lds_off;                                         // lds_addr (bytes)
  d.g0[2] = (unsigned)(ga & 0xFFFFFFFFu);                    // global_addr[31:0]
  d.g0[3] = (unsigned)((ga >> 32) & 0x1FFFFFFu) | (2u << 30); // addr[56:32] | type=2
  unsigned f0 = (1u << 16);                                  // data_size = 2 bytes
  if (pad_en) f0 |= (1u << 20) | (pad_int_code << 22) | (pad_amt_code << 25);
  d.g1[0] = (int)f0;                                         // wg_mask=0, flags
  d.g1[1] = (int)((tensor_d0 & 0xFFFFu) << 16);              // tensor_dim0[15:0] @63:48
  d.g1[2] = (int)(((tensor_d0 >> 16) & 0xFFFFu) | ((tensor_d1 & 0xFFFFu) << 16));
  d.g1[3] = (int)(((tensor_d1 >> 16) & 0xFFFFu) | ((tile_d0 & 0xFFFFu) << 16));
  d.g1[4] = (int)(tile_d1 & 0xFFFFu);                        // tile_dim1, tile_dim2=0
  d.g1[5] = (int)stride_d0;                                  // tensor_dim0_stride[31:0]
  d.g1[6] = 0;
  d.g1[7] = 0;
  return d;
}

__device__ __forceinline__ void tdm_issue(const TdmDesc& d) {
  v4i g2 = {0, 0, 0, 0};
  v4i g3 = {0, 0, 0, 0};
#if __clang_major__ >= 23
  v8i g4 = {0, 0, 0, 0, 0, 0, 0, 0};
  __builtin_amdgcn_tensor_load_to_lds(d.g0, d.g1, g2, g3, g4, 0);
#else
  __builtin_amdgcn_tensor_load_to_lds(d.g0, d.g1, g2, g3, 0);
#endif
}
#endif

// ---------------- fp32 -> bf16 convert (4-wide) ----------------
__global__ __launch_bounds__(256) void conv_bf16_4(const float* __restrict__ in,
                                                   __bf16* __restrict__ out, long n4) {
  long i = (long)blockIdx.x * 256 + threadIdx.x;
  if (i >= n4) return;
  float4 f = ((const float4*)in)[i];
  __bf16* o = out + i * 4;
  o[0] = (__bf16)f.x; o[1] = (__bf16)f.y; o[2] = (__bf16)f.z; o[3] = (__bf16)f.w;
}

// ---------------- transpose + convert: W[K][N] fp32 -> Wt[N][K] bf16 ----------------
__global__ __launch_bounds__(256) void transconv(const float* __restrict__ W,
                                                 __bf16* __restrict__ Wt,
                                                 int Kdim, int Ndim) {
  __shared__ float tile[32][33];
  int nb = blockIdx.x * 32, kb = blockIdx.y * 32;
  int tx = threadIdx.x, ty = threadIdx.y;               // 32 x 8
#pragma unroll
  for (int i = 0; i < 4; ++i)
    tile[ty + i * 8][tx] = W[(size_t)(kb + ty + i * 8) * Ndim + nb + tx];
  __syncthreads();
#pragma unroll
  for (int i = 0; i < 4; ++i)
    Wt[(size_t)(nb + ty + i * 8) * Kdim + kb + tx] = (__bf16)tile[tx][ty + i * 8];
}

// ---------------- GEMM core: C(128x128) += A(M,K) x Bt(N,K)^T, bf16 WMMA ----------------
// TDM path: double-buffered LDS, descriptors built uniformly, wave0 drives the DMA.
__device__ __forceinline__ void gemm_core(const __bf16* __restrict__ A,
                                          const __bf16* __restrict__ Bt,
                                          int Kdim, int m0, int n0,
                                          char* lds, v8f acc[2][4]) {
  const int tid = threadIdx.x, lane = tid & 31, wave = tid >> 5;
  const int wm = wave & 3, wn = wave >> 2;
  const int rsel = lane & 15, ksel = (lane >> 4) * 8;
#if HAVE_TDM
  const int nIter = Kdim / 32;
  {
    // tile rows: 32 bf16 = 64B, pad 32B -> LDT*2 = 96B stride.
    // pad_interval code 3 = 16 DWORDs (64B); pad_amount code 7 = 8 DWORDs (32B).
    TdmDesc dA = tdm_make_2d(0u, &A[(size_t)m0 * Kdim], 32u, 128u,
                             (unsigned)Kdim, 128u, (unsigned)Kdim, 3u, 7u, true);
    TdmDesc dB = tdm_make_2d(12288u, &Bt[(size_t)n0 * Kdim], 32u, 128u,
                             (unsigned)Kdim, 128u, (unsigned)Kdim, 3u, 7u, true);
    if (wave == 0) { tdm_issue(dA); tdm_issue(dB); }
  }
  for (int it = 0; it < nIter; ++it) {
    const unsigned base = (unsigned)(it & 1) * GEMM_BUF;
    if (it + 1 < nIter) {                 // uniform branch (loop counter)
      const int kk = (it + 1) * 32;
      const unsigned nb = (unsigned)((it + 1) & 1) * GEMM_BUF;
      TdmDesc dA = tdm_make_2d(nb, &A[(size_t)m0 * Kdim + kk], 32u, 128u,
                               (unsigned)(Kdim - kk), 128u, (unsigned)Kdim, 3u, 7u, true);
      TdmDesc dB = tdm_make_2d(nb + 12288u, &Bt[(size_t)n0 * Kdim + kk], 32u, 128u,
                               (unsigned)(Kdim - kk), 128u, (unsigned)Kdim, 3u, 7u, true);
      if (wave == 0) {
        tdm_issue(dA);
        tdm_issue(dB);
        __builtin_amdgcn_s_wait_tensorcnt((short)2);  // current buffer's loads done
      }
    } else {
      if (wave == 0) __builtin_amdgcn_s_wait_tensorcnt((short)0);
    }
    __syncthreads();
    __bf16* Ash = (__bf16*)(lds + base);
    __bf16* Bsh = Ash + 128 * LDT;
    v16bf af[2], bfr[4];
#pragma unroll
    for (int mt = 0; mt < 2; ++mt) {
      const __bf16* p = &Ash[(wm * 32 + mt * 16 + rsel) * LDT + ksel];
      af[mt] = ldfrag(p, p + 16);
    }
#pragma unroll
    for (int nt = 0; nt < 4; ++nt) {
      const __bf16* p = &Bsh[(wn * 64 + nt * 16 + rsel) * LDT + ksel];
      bfr[nt] = ldfrag(p, p + 16);
    }
#pragma unroll
    for (int mt = 0; mt < 2; ++mt)
#pragma unroll
      for (int nt = 0; nt < 4; ++nt)
        acc[mt][nt] = wmma_bf16(af[mt], bfr[nt], acc[mt][nt]);
    __syncthreads();   // all reads of buf done before it is overwritten next iter
  }
#else
  __bf16* Ash = (__bf16*)lds;
  __bf16* Bsh = Ash + 128 * LDT;
  for (int kk = 0; kk < Kdim; kk += 32) {
#pragma unroll
    for (int i = 0; i < 2; ++i) {
      int c = tid + i * 256;
      int row = c >> 2, kc = (c & 3) * 8;
      *(v8bf*)&Ash[row * LDT + kc] = *(const v8bf*)&A[(size_t)(m0 + row) * Kdim + kk + kc];
      *(v8bf*)&Bsh[row * LDT + kc] = *(const v8bf*)&Bt[(size_t)(n0 + row) * Kdim + kk + kc];
    }
    __syncthreads();
    v16bf af[2], bfr[4];
#pragma unroll
    for (int mt = 0; mt < 2; ++mt) {
      const __bf16* p = &Ash[(wm * 32 + mt * 16 + rsel) * LDT + ksel];
      af[mt] = ldfrag(p, p + 16);
    }
#pragma unroll
    for (int nt = 0; nt < 4; ++nt) {
      const __bf16* p = &Bsh[(wn * 64 + nt * 16 + rsel) * LDT + ksel];
      bfr[nt] = ldfrag(p, p + 16);
    }
#pragma unroll
    for (int mt = 0; mt < 2; ++mt)
#pragma unroll
      for (int nt = 0; nt < 4; ++nt)
        acc[mt][nt] = wmma_bf16(af[mt], bfr[nt], acc[mt][nt]);
    __syncthreads();
  }
#endif
}

// ---------------- GEMM 1: qkv = xbf @ w_attn, scatter to head-major q/k/v bf16 ----------------
__global__ __launch_bounds__(256) void gemm_qkv_k(const __bf16* __restrict__ A,
                                                  const __bf16* __restrict__ Bt,
                                                  __bf16* __restrict__ qh,
                                                  __bf16* __restrict__ kh,
                                                  __bf16* __restrict__ vh) {
  extern __shared__ __align__(16) char smem[];
  v8f acc[2][4];
#pragma unroll
  for (int mt = 0; mt < 2; ++mt)
#pragma unroll
    for (int nt = 0; nt < 4; ++nt)
#pragma unroll
      for (int r = 0; r < 8; ++r) acc[mt][nt][r] = 0.0f;

  int m0 = blockIdx.x * 128, n0 = blockIdx.y * 128;
  gemm_core(A, Bt, N_C, m0, n0, smem, acc);

  const int lane = threadIdx.x & 31, wave = threadIdx.x >> 5;
  const int wm = wave & 3, wn = wave >> 2;
#pragma unroll
  for (int mt = 0; mt < 2; ++mt)
#pragma unroll
    for (int nt = 0; nt < 4; ++nt)
#pragma unroll
      for (int r = 0; r < 8; ++r) {
        int m = m0 + wm * 32 + mt * 16 + r + 8 * (lane >> 4);
        int n = n0 + wn * 64 + nt * 16 + (lane & 15);
        int which = n >> 11;           // 0=q 1=k 2=v
        int c = n & 2047;
        int h = c >> 7, d = c & 127;
        int b = m >> 11, t = m & 2047; // T = 2048
        __bf16* dst = (which == 0) ? qh : (which == 1) ? kh : vh;
        dst[(((size_t)b * N_H + h) * N_T + t) * N_HD + d] = (__bf16)acc[mt][nt][r];
      }
}

// ---------------- GEMM 2: out = aout @ w_proj, fp32 store ----------------
__global__ __launch_bounds__(256) void gemm_proj_k(const __bf16* __restrict__ A,
                                                   const __bf16* __restrict__ Bt,
                                                   float* __restrict__ out) {
  extern __shared__ __align__(16) char smem[];
  v8f acc[2][4];
#pragma unroll
  for (int mt = 0; mt < 2; ++mt)
#pragma unroll
    for (int nt = 0; nt < 4; ++nt)
#pragma unroll
      for (int r = 0; r < 8; ++r) acc[mt][nt][r] = 0.0f;

  int m0 = blockIdx.x * 128, n0 = blockIdx.y * 128;
  gemm_core(A, Bt, N_C, m0, n0, smem, acc);

  const int lane = threadIdx.x & 31, wave = threadIdx.x >> 5;
  const int wm = wave & 3, wn = wave >> 2;
#pragma unroll
  for (int mt = 0; mt < 2; ++mt)
#pragma unroll
    for (int nt = 0; nt < 4; ++nt)
#pragma unroll
      for (int r = 0; r < 8; ++r) {
        int m = m0 + wm * 32 + mt * 16 + r + 8 * (lane >> 4);
        int n = n0 + wn * 64 + nt * 16 + (lane & 15);
        out[(size_t)m * N_C + n] = acc[mt][nt][r];
      }
}

// ---------------- RoPE (in place, bf16 head-major buffer) ----------------
__global__ __launch_bounds__(256) void rope_k(__bf16* __restrict__ buf,
                                              const float* __restrict__ cs,
                                              const float* __restrict__ sn) {
  long idx = (long)blockIdx.x * 256 + threadIdx.x;
  long total = (long)N_B * N_H * N_T * (N_HD / 2);
  if (idx >= total) return;
  int i = (int)(idx & 63);
  long rest = idx >> 6;
  int t = (int)(rest & (N_T - 1));
  long bh = rest >> 11;
  float c = cs[t * 64 + i], s = sn[t * 64 + i];
  __bf16* p = buf + ((size_t)bh * N_T + t) * N_HD + 2 * i;
  float x1 = (float)p[0], x2 = (float)p[1];
  p[0] = (__bf16)(x1 * c - x2 * s);
  p[1] = (__bf16)(x1 * s + x2 * c);
}

// ---------------- Flash attention: per (b,h), 128-query block, causal ----------------
__global__ __launch_bounds__(256) void flash_k(const __bf16* __restrict__ Q,
                                               const __bf16* __restrict__ Kb,
                                               const __bf16* __restrict__ Vb,
                                               __bf16* __restrict__ Ob) {
  const int bh = blockIdx.y;
  const int blk = blockIdx.x;
  const int b = bh >> 4, h = bh & 15;
  const __bf16* q = Q + (size_t)bh * N_T * N_HD;
  const __bf16* kg = Kb + (size_t)bh * N_T * N_HD;
  const __bf16* vg = Vb + (size_t)bh * N_T * N_HD;

  const int tid = threadIdx.x, lane = tid & 31, wave = tid >> 5;
  const int rsel = lane & 15, ksel = (lane >> 4) * 8;
  const int q0 = blk * 128 + wave * 16;
  const float scale = 0.08838834764831845f;   // 1/sqrt(128)

  extern __shared__ __align__(16) char smem[];
  __bf16* Ksh = (__bf16*)smem;                    // 64 x KST (LDS offset 0)
  __bf16* Vt  = Ksh + 64 * KST;                   // 128 x VST (transposed V)
  __bf16* Psh = Vt + 128 * VST + wave * 16 * 32;  // per-wave 16x32 staging

  v16bf qf[4];
  {
    const __bf16* qrow = q + (size_t)(q0 + rsel) * N_HD;
#pragma unroll
    for (int c = 0; c < 4; ++c)
      qf[c] = ldfrag(qrow + c * 32 + ksel, qrow + c * 32 + ksel + 16);
  }

  v8f o[8];
  float mrow[8], lrow[8];
#pragma unroll
  for (int nt = 0; nt < 8; ++nt)
#pragma unroll
    for (int r = 0; r < 8; ++r) o[nt][r] = 0.0f;
#pragma unroll
  for (int r = 0; r < 8; ++r) { mrow[r] = -INFINITY; lrow[r] = 0.0f; }

  const int nch = (blk + 1) * 2;   // 64-key chunks up to causal frontier
  for (int ch = 0; ch < nch; ++ch) {
    const int j0 = ch * 64;
#if HAVE_TDM
    {
      // K rows 256B + 16B pad -> KST stride; interval code 5 (64 DW), amount code 3 (4 DW).
      TdmDesc dK = tdm_make_2d(0u, &kg[(size_t)j0 * N_HD], (unsigned)N_HD, 64u,
                               (unsigned)N_HD, (unsigned)(N_T - j0), (unsigned)N_HD,
                               5u, 3u, true);
      if (wave == 0) tdm_issue(dK);
    }
#else
#pragma unroll
    for (int i = 0; i < 2; ++i) {
      int c = tid + i * 256;
      int row = c >> 3, kc = (c & 7) * 8;
      *(v8bf*)&Ksh[row * KST + kc] = *(const v8bf*)&kg[(size_t)(j0 + row) * N_HD + kc];
    }
#endif
#pragma unroll
    for (int i = 0; i < 4; ++i) {           // V chunk transposed: Vt[d][key]
      int c = tid + i * 256;
      int row = c >> 4, d0 = (c & 15) * 8;
      v8bf vv = *(const v8bf*)&vg[(size_t)(j0 + row) * N_HD + d0];
#pragma unroll
      for (int e = 0; e < 8; ++e) Vt[(d0 + e) * VST + row] = vv[e];
    }
#if HAVE_TDM
    if (wave == 0) __builtin_amdgcn_s_wait_tensorcnt((short)0);
#endif
    __syncthreads();

#pragma unroll
    for (int sub = 0; sub < 2; ++sub) {
      const int kbase = sub * 32;
      v8f s0, s1;
#pragma unroll
      for (int r = 0; r < 8; ++r) { s0[r] = 0.0f; s1[r] = 0.0f; }
#pragma unroll
      for (int c = 0; c < 4; ++c) {
        const __bf16* p0 = &Ksh[(kbase + rsel) * KST + c * 32 + ksel];
        const __bf16* p1 = &Ksh[(kbase + 16 + rsel) * KST + c * 32 + ksel];
        v16bf k0 = ldfrag(p0, p0 + 16);
        v16bf k1 = ldfrag(p1, p1 + 16);
        s0 = wmma_bf16(qf[c], k0, s0);
        s1 = wmma_bf16(qf[c], k1, s1);
      }
      const int key0 = j0 + kbase + (lane & 15);
      const int rowg0 = q0 + 8 * (lane >> 4);
      float rmax[8];
#pragma unroll
      for (int r = 0; r < 8; ++r) {
        float a = s0[r] * scale;
        float bb = s1[r] * scale;
        int rowg = rowg0 + r;
        if (key0 > rowg) a = -INFINITY;
        if (key0 + 16 > rowg) bb = -INFINITY;
        s0[r] = a; s1[r] = bb;
        float loc = fmaxf(a, bb);
#pragma unroll
        for (int off = 1; off < 16; off <<= 1)
          loc = fmaxf(loc, __shfl_xor(loc, off, 32));
        rmax[r] = loc;
      }
#pragma unroll
      for (int r = 0; r < 8; ++r) {
        float mnew = fmaxf(mrow[r], rmax[r]);
        float corr = __expf(mrow[r] - mnew);
        float p0 = __expf(s0[r] - mnew);
        float p1 = __expf(s1[r] - mnew);
        float rs = p0 + p1;
#pragma unroll
        for (int off = 1; off < 16; off <<= 1)
          rs += __shfl_xor(rs, off, 32);
        lrow[r] = lrow[r] * corr + rs;
        mrow[r] = mnew;
#pragma unroll
        for (int nt = 0; nt < 8; ++nt) o[nt][r] *= corr;
        int prow = r + 8 * (lane >> 4);
        Psh[prow * 32 + (lane & 15)] = (__bf16)p0;
        Psh[prow * 32 + 16 + (lane & 15)] = (__bf16)p1;
      }
      asm volatile("s_wait_dscnt 0" ::: "memory");  // wave-local LDS store->load ordering
      v16bf pf = ldfrag(&Psh[rsel * 32 + ksel], &Psh[rsel * 32 + ksel + 16]);
#pragma unroll
      for (int nt = 0; nt < 8; ++nt) {
        const __bf16* pv = &Vt[(nt * 16 + rsel) * VST + kbase + ksel];
        v16bf vf = ldfrag(pv, pv + 16);
        o[nt] = wmma_bf16(pf, vf, o[nt]);
      }
    }
    __syncthreads();
  }

  float inv[8];
#pragma unroll
  for (int r = 0; r < 8; ++r) inv[r] = 1.0f / lrow[r];
#pragma unroll
  for (int nt = 0; nt < 8; ++nt)
#pragma unroll
    for (int r = 0; r < 8; ++r) {
      int trow = q0 + r + 8 * (lane >> 4);
      int d = nt * 16 + (lane & 15);
      Ob[((size_t)b * N_T + trow) * N_C + h * N_HD + d] = (__bf16)(o[nt][r] * inv[r]);
    }
}

// ---------------- Host launcher ----------------
extern "C" void kernel_launch(void* const* d_in, const int* in_sizes, int n_in,
                              void* d_out, int out_size, void* d_ws, size_t ws_size,
                              hipStream_t stream) {
  (void)in_sizes; (void)n_in; (void)out_size; (void)ws_size;
  const float* x      = (const float*)d_in[0];
  const float* w_attn = (const float*)d_in[1];
  const float* w_proj = (const float*)d_in[2];
  const float* rc     = (const float*)d_in[3];
  const float* rs     = (const float*)d_in[4];
  float* out = (float*)d_out;

  const long M1 = (long)N_B * N_T;   // 8192
  const int  N1 = 3 * N_C;           // 6144

  char* ws = (char*)d_ws;
  size_t off = 0;
  auto take = [&](size_t bytes) -> char* {
    char* p = ws + off;
    off = (off + bytes + 255) & ~(size_t)255;
    return p;
  };
  __bf16* xbf  = (__bf16*)take((size_t)M1 * N_C * 2);
  __bf16* waT  = (__bf16*)take((size_t)N1 * N_C * 2);
  __bf16* wpT  = (__bf16*)take((size_t)N_C * N_C * 2);
  __bf16* qh   = (__bf16*)take((size_t)N_B * N_H * N_T * N_HD * 2);
  __bf16* kh   = (__bf16*)take((size_t)N_B * N_H * N_T * N_HD * 2);
  __bf16* vh   = (__bf16*)take((size_t)N_B * N_H * N_T * N_HD * 2);
  __bf16* aout = (__bf16*)take((size_t)M1 * N_C * 2);

  {
    long n4 = M1 * N_C / 4;
    conv_bf16_4<<<(unsigned)((n4 + 255) / 256), 256, 0, stream>>>(x, xbf, n4);
  }
  transconv<<<dim3(N1 / 32, N_C / 32), dim3(32, 8), 0, stream>>>(w_attn, waT, N_C, N1);
  transconv<<<dim3(N_C / 32, N_C / 32), dim3(32, 8), 0, stream>>>(w_proj, wpT, N_C, N_C);

#if HAVE_TDM
  const size_t gemm_smem = 2u * GEMM_BUF;                 // double-buffered: 49152 B
#else
  const size_t gemm_smem = GEMM_BUF;                      // single buffer: 24576 B
#endif
  gemm_qkv_k<<<dim3((unsigned)(M1 / 128), N1 / 128), 256, gemm_smem, stream>>>(
      xbf, waT, qh, kh, vh);

  {
    long total = (long)N_B * N_H * N_T * (N_HD / 2);
    unsigned blocks = (unsigned)((total + 255) / 256);
    rope_k<<<blocks, 256, 0, stream>>>(qh, rc, rs);
    rope_k<<<blocks, 256, 0, stream>>>(kh, rc, rs);
  }

  {
    const size_t fl_smem = (size_t)64 * KST * 2 + (size_t)128 * VST * 2 + (size_t)8 * 16 * 32 * 2;
    flash_k<<<dim3(N_T / 128, N_B * N_H), 256, fl_smem, stream>>>(qh, kh, vh, aout);
  }

  gemm_proj_k<<<dim3((unsigned)(M1 / 128), N_C / 128), 256, gemm_smem, stream>>>(
      aout, wpT, out);
}
